// SSM_74045236183427
// MI455X (gfx1250) — matrix operand order repaired and checked by reading, not compiled, not run on hardware
//
#include <hip/hip_runtime.h>
#include <hip/hip_bf16.h>
#include <math.h>

// ---------------------------------------------------------------------------
// Problem constants (from reference)
// ---------------------------------------------------------------------------
constexpr int S_   = 2048;
constexpr int E_   = 2048;
constexpr int NH   = 64;
constexpr int P_   = 64;
constexpr int N_   = 128;
constexpr int KW   = 4;          // conv kernel width
constexpr int CS   = 256;        // chunk size
constexpr int INTER    = NH * P_;               // 4096
constexpr int CONV_DIM = INTER + 2 * N_;        // 4352  (G=1)
constexpr int PROJ     = INTER + CONV_DIM + NH; // 8512
constexpr int NCHUNK   = S_ / CS;               // 8
constexpr float EPS = 1e-5f;

typedef __bf16 bf16_t;
typedef bf16_t bf16x16 __attribute__((ext_vector_type(16)));
typedef bf16_t bf16x8  __attribute__((ext_vector_type(8)));
typedef bf16_t bf16x4  __attribute__((ext_vector_type(4)));
typedef float  v8f     __attribute__((ext_vector_type(8)));

#define DEVFN static __device__ __forceinline__
#define AS1 __attribute__((address_space(1)))
#define AS3 __attribute__((address_space(3)))

#if defined(__gfx1250__) && __has_builtin(__builtin_amdgcn_global_load_async_to_lds_b128)
#define HAVE_ASYNC_LDS 1
#else
#define HAVE_ASYNC_LDS 0
#endif

#if HAVE_ASYNC_LDS
// 16B global -> LDS async copy (ASYNCcnt). Builtin signature (from probe):
// (int4 AS1*, int4 AS3*, imm offset, imm cpol). Address-space casts go via
// uintptr_t: flat LDS addresses carry the LDS byte offset in the low 32 bits.
typedef int v4i_vs __attribute__((vector_size(16)));
DEVFN void async_copy_b128(const void* g, void* l) {
  __builtin_amdgcn_global_load_async_to_lds_b128(
      (AS1 v4i_vs*)(uintptr_t)g, (AS3 v4i_vs*)(uintptr_t)l, 0, 0);
}
DEVFN void wait_async0() {
#if __has_builtin(__builtin_amdgcn_s_wait_asynccnt)
  __builtin_amdgcn_s_wait_asynccnt(0);
#else
  asm volatile("s_wait_asynccnt 0x0" ::: "memory");
#endif
}
#endif

// ---------------------------------------------------------------------------
// WMMA helpers (CDNA5: D = A(16x32 bf16) x B(32x16 bf16) + C(16x16 f32))
// ---------------------------------------------------------------------------
DEVFN v8f wmma_bf16(bf16x16 a, bf16x16 b, v8f c) {
  return __builtin_amdgcn_wmma_f32_16x16x32_bf16(false, a, false, b,
                                                 (short)0, c, false, false);
}

// A-matrix fragment (16x32, row-major in LDS). Per ISA 7.12.2:
// lanes 0-15: row=lane,     K = 0..7  and 16..23
// lanes 16-31: row=lane-16, K = 8..15 and 24..31
DEVFN bf16x16 load_frag_a(const bf16_t* base, int row, int stride, int lane) {
  const bf16_t* p = base + row * stride + ((lane < 16) ? 0 : 8);
  bf16x16 f;
#pragma unroll
  for (int i = 0; i < 8; ++i) f[i] = p[i];
#pragma unroll
  for (int i = 0; i < 8; ++i) f[8 + i] = p[16 + i];
  return f;
}

// B-matrix fragment (32x16). LDS holds it N-major (row = column n, K contig).
// lanes 0-15: col=lane, K=0..7 and 16..23 ; lanes 16-31: col=lane-16, K=8..15 and 24..31
DEVFN bf16x16 load_frag_b(const bf16_t* base, int col, int stride, int lane) {
  const bf16_t* p = base + col * stride + ((lane < 16) ? 0 : 8);
  bf16x16 f;
#pragma unroll
  for (int i = 0; i < 8; ++i) f[i] = p[i];
#pragma unroll
  for (int i = 0; i < 8; ++i) f[8 + i] = p[16 + i];
  return f;
}

DEVFN bf16x4 cvt4(float4 v) {
  bf16x4 r = {(bf16_t)v.x, (bf16_t)v.y, (bf16_t)v.z, (bf16_t)v.w};
  return r;
}

// ---------------------------------------------------------------------------
// NT GEMM  C[M,N] = A[M,K] * B[N,K]^T  (fp32 in/out, bf16 WMMA, fp32 acc).
// Block tile 256x64, 8 waves, software-pipelined staging.
// Requires M%256==0, N%64==0, K%32==0 (true for all uses here).
// ---------------------------------------------------------------------------
__global__ __launch_bounds__(256) void gemm_nt_wmma(
    const float* __restrict__ A, const float* __restrict__ B,
    float* __restrict__ C, int M, int N, int K) {
  __shared__ __align__(16) bf16_t As[256][40];
  __shared__ __align__(16) bf16_t Bs[64][40];
  const int tid  = threadIdx.x;
  const int lane = tid & 31;
  const int wave = tid >> 5;
  const int m0 = blockIdx.y * 256;
  const int n0 = blockIdx.x * 64;

  const int ar = tid >> 3;        // 0..31
  const int ac = (tid & 7) * 4;   // 0,4,..,28
  const float* aptr = A + (size_t)(m0 + ar) * K + ac;
  const float* bptr = B + (size_t)(n0 + ar) * K + ac;

  float4 ra[8], rb[2];
  auto issue_loads = [&](int kk) {
#pragma unroll
    for (int i = 0; i < 8; ++i)
      ra[i] = *(const float4*)(aptr + (size_t)i * 32 * K + kk);
#pragma unroll
    for (int i = 0; i < 2; ++i)
      rb[i] = *(const float4*)(bptr + (size_t)i * 32 * K + kk);
  };

  v8f acc[2][4] = {};
  issue_loads(0);

  for (int kk = 0; kk < K; kk += 32) {
    __syncthreads();  // previous tile's LDS readers done
#pragma unroll
    for (int i = 0; i < 8; ++i)
      *(bf16x4*)&As[ar + i * 32][ac] = cvt4(ra[i]);
#pragma unroll
    for (int i = 0; i < 2; ++i)
      *(bf16x4*)&Bs[ar + i * 32][ac] = cvt4(rb[i]);
    __syncthreads();

    if (kk + 32 < K) issue_loads(kk + 32);          // overlap with WMMA
    if (kk + 64 < K) __builtin_prefetch(aptr + kk + 64, 0, 1);

    bf16x16 a0 = load_frag_a(&As[0][0], wave * 16 + (lane & 15), 40, lane);
    bf16x16 a1 = load_frag_a(&As[0][0], 128 + wave * 16 + (lane & 15), 40, lane);
#pragma unroll
    for (int j = 0; j < 4; ++j) {
      bf16x16 bfrag = load_frag_b(&Bs[0][0], j * 16 + (lane & 15), 40, lane);
      acc[0][j] = wmma_bf16(a0, bfrag, acc[0][j]);
      acc[1][j] = wmma_bf16(a1, bfrag, acc[1][j]);
    }
  }

  const int colloc = lane & 15;
#pragma unroll
  for (int g = 0; g < 2; ++g) {
#pragma unroll
    for (int j = 0; j < 4; ++j) {
#pragma unroll
      for (int i = 0; i < 8; ++i) {
        int row = m0 + g * 128 + wave * 16 + ((lane < 16) ? i : i + 8);
        int col = n0 + j * 16 + colloc;
        C[(size_t)row * N + col] = acc[g][j][i];
      }
    }
  }
}

// ---------------------------------------------------------------------------
// Depthwise causal conv (K=4) + SiLU. Output stored as bf16 (WMMA consumer).
// ---------------------------------------------------------------------------
__global__ __launch_bounds__(256) void conv_silu_kernel(
    const float* __restrict__ proj, const float* __restrict__ conv_w,
    const float* __restrict__ conv_b, bf16_t* __restrict__ out) {
  constexpr int CD4 = CONV_DIM / 4;  // 1088
  int idx = blockIdx.x * 256 + threadIdx.x;
  if (idx >= S_ * CD4) return;
  int s = idx / CD4;
  int c4 = (idx - s * CD4) * 4;

  float4 b4 = *(const float4*)&conv_b[c4];
  float acc[4] = {b4.x, b4.y, b4.z, b4.w};
  float4 w[4];
#pragma unroll
  for (int j = 0; j < 4; ++j) w[j] = *(const float4*)&conv_w[(c4 + j) * KW];

#pragma unroll
  for (int i = 0; i < KW; ++i) {
    int sp = s - (KW - 1) + i;
    if (sp >= 0) {
      float4 x = *(const float4*)&proj[(size_t)sp * PROJ + INTER + c4];
      acc[0] += x.x * ((const float*)&w[0])[i];
      acc[1] += x.y * ((const float*)&w[1])[i];
      acc[2] += x.z * ((const float*)&w[2])[i];
      acc[3] += x.w * ((const float*)&w[3])[i];
    }
  }
  bf16x4 o;
#pragma unroll
  for (int j = 0; j < 4; ++j)
    o[j] = (bf16_t)(acc[j] / (1.f + __expf(-acc[j])));
  *(bf16x4*)&out[(size_t)s * CONV_DIM + c4] = o;
}

// ---------------------------------------------------------------------------
// dt = softplus(dt_raw + bias); dA = -exp(A_log)*dt
// ---------------------------------------------------------------------------
__global__ __launch_bounds__(256) void dt_kernel(
    const float* __restrict__ proj, const float* __restrict__ dt_bias,
    const float* __restrict__ A_log, float* __restrict__ dtsp,
    float* __restrict__ dAv) {
  int idx = blockIdx.x * 256 + threadIdx.x;
  if (idx >= S_ * NH) return;
  int s = idx >> 6, hh = idx & 63;
  float x = proj[(size_t)s * PROJ + INTER + CONV_DIM + hh] + dt_bias[hh];
  float sp = (x > 20.f) ? x : log1pf(__expf(x));  // softplus; clip(.,0) no-op
  dtsp[idx] = sp;
  dAv[idx]  = -__expf(A_log[hh]) * sp;
}

// ---------------------------------------------------------------------------
// Chunked SSD. Block = (l-tile of 64 rows, head, chunk).
// Y[l,p] = sum_s Gm[l,s] * W(l,s) * dt[s] * H[s,p] + D[h]*H[l,p]
//   Gm = C·B^T (K=128, WMMA);  W = [s<=l] e^{a_l-a_s} + e^{a_l+a_end-a_s}
// dt[s] is folded into the mask so B/C/H tiles are raw bf16 copies
// (async-to-LDS when available).
// ---------------------------------------------------------------------------
__global__ __launch_bounds__(256) void ssd_kernel(
    const bf16_t* __restrict__ convb, const float* __restrict__ dtsp,
    const float* __restrict__ dAv, const float* __restrict__ Dvec,
    float* __restrict__ y) {
  const int lt = blockIdx.x;        // 0..3
  const int h  = blockIdx.y;        // 0..63
  const int c  = blockIdx.z;        // 0..7
  const int tid = threadIdx.x, lane = tid & 31, wave = tid >> 5;
  const int s_chunk = c * CS;
  const int l0 = lt * 64;
  const int tr = wave >> 1;         // wave's 16-row tile within 64

  __shared__ float  a_sh[CS];
  __shared__ float  dt_sh[CS];
  __shared__ __align__(16) bf16_t Cl[64][136];
  __shared__ __align__(16) bf16_t Bt[64][136];
  __shared__ __align__(16) bf16_t Ht[64][72];   // transposed: [p][s_local]
  __shared__ __align__(16) bf16_t Mt[64][72];

  // Stage C rows for this l-tile (64 x 128) -- raw bf16 16B-segment copy
#pragma unroll
  for (int i = 0; i < 4; ++i) {
    int seg = i * 256 + tid;
    int r = seg >> 4, cc = (seg & 15) * 8;
    const bf16_t* gp =
        &convb[(size_t)(s_chunk + l0 + r) * CONV_DIM + INTER + N_ + cc];
#if HAVE_ASYNC_LDS
    async_copy_b128(gp, &Cl[r][cc]);
#else
    *(uint4*)&Cl[r][cc] = *(const uint4*)gp;
#endif
  }

  // Inclusive scan of dA over the chunk (Hillis-Steele in LDS) + dt values
  a_sh[tid]  = dAv[(size_t)(s_chunk + tid) * NH + h];
  dt_sh[tid] = dtsp[(size_t)(s_chunk + tid) * NH + h];
  __syncthreads();
  for (int off = 1; off < CS; off <<= 1) {
    float v = (tid >= off) ? a_sh[tid - off] : 0.f;
    __syncthreads();
    a_sh[tid] += v;
    __syncthreads();
  }
  const float a_end = a_sh[CS - 1];

  v8f yacc[2] = {};
  const float Dh = Dvec[h];

  for (int st = 0; st < 4; ++st) {
    const int s0 = st * 64;
    // Stage B tile (64 x 128) -- raw bf16 copy
#pragma unroll
    for (int i = 0; i < 4; ++i) {
      int seg = i * 256 + tid;
      int r = seg >> 4, cc = (seg & 15) * 8;
      const bf16_t* gp =
          &convb[(size_t)(s_chunk + s0 + r) * CONV_DIM + INTER + cc];
#if HAVE_ASYNC_LDS
      async_copy_b128(gp, &Bt[r][cc]);
#else
      *(uint4*)&Bt[r][cc] = *(const uint4*)gp;
#endif
    }
    // Stage H tile transposed (p-major): Ht[p][s] = H[s,p]  (raw, dt folded
    // into M); vector global load, scalar LDS transpose stores
    {
      const int r0 = tid >> 3;        // 0..31 (s row)
      const int p0 = (tid & 7) * 8;   // 0..56
#pragma unroll
      for (int i = 0; i < 2; ++i) {
        int sl = r0 + i * 32;
        int sg = s_chunk + s0 + sl;
        bf16x8 v = *(const bf16x8*)&convb[(size_t)sg * CONV_DIM + h * P_ + p0];
#pragma unroll
        for (int j = 0; j < 8; ++j) Ht[p0 + j][sl] = v[j];
      }
    }
#if HAVE_ASYNC_LDS
    wait_async0();
#endif
    __syncthreads();

    // Phase 1: Gm tile (64x64) = Cl(64x128) x Bt^T ; mask*dt -> bf16 Mt
    {
      v8f g[2] = {};
#pragma unroll
      for (int kk = 0; kk < 4; ++kk) {
        bf16x16 a = load_frag_a(&Cl[0][kk * 32], tr * 16 + (lane & 15), 136, lane);
#pragma unroll
        for (int t = 0; t < 2; ++t) {
          int tc = (wave & 1) * 2 + t;
          bf16x16 b = load_frag_b(&Bt[0][kk * 32], tc * 16 + (lane & 15), 136, lane);
          g[t] = wmma_bf16(a, b, g[t]);
        }
      }
      const int colw = lane & 15;
#pragma unroll
      for (int t = 0; t < 2; ++t) {
        const int tc = (wave & 1) * 2 + t;
#pragma unroll
        for (int i = 0; i < 8; ++i) {
          int row = (lane < 16) ? i : i + 8;
          int l_loc = l0 + tr * 16 + row;
          int s_loc = s0 + tc * 16 + colw;
          float el = a_sh[l_loc], es = a_sh[s_loc];
          float w = __expf(el + a_end - es);         // "Yoff" term (exp <= 0)
          if (s_loc <= l_loc) w += __expf(el - es);  // causal intra-chunk term
          Mt[tr * 16 + row][tc * 16 + colw] =
              (bf16_t)(g[t][i] * w * dt_sh[s_loc]);
        }
      }
    }
    __syncthreads();

    // Phase 2: Y(64x64) += Mt(64x64) x H(64x64)
#pragma unroll
    for (int kk = 0; kk < 2; ++kk) {
      bf16x16 a = load_frag_a(&Mt[0][kk * 32], tr * 16 + (lane & 15), 72, lane);
#pragma unroll
      for (int t = 0; t < 2; ++t) {
        int tc = (wave & 1) * 2 + t;
        bf16x16 b = load_frag_b(&Ht[0][kk * 32], tc * 16 + (lane & 15), 72, lane);
        yacc[t] = wmma_bf16(a, b, yacc[t]);
      }
    }
    __syncthreads();
  }

  // Epilogue: add D residual (hidden WITHOUT dt) and store
  const int colw = lane & 15;
#pragma unroll
  for (int t = 0; t < 2; ++t) {
    const int tc = (wave & 1) * 2 + t;
#pragma unroll
    for (int i = 0; i < 8; ++i) {
      int row = (lane < 16) ? i : i + 8;
      int sg = s_chunk + l0 + tr * 16 + row;
      int p = tc * 16 + colw;
      float hraw = (float)convb[(size_t)sg * CONV_DIM + h * P_ + p];
      y[(size_t)sg * INTER + h * P_ + p] = yacc[t][i] + Dh * hraw;
    }
  }
}

// ---------------------------------------------------------------------------
// Gated RMSNorm:  yn = norm_w * (y*silu(gate)) * rsqrt(mean+eps)
// One block per sequence position; in-place on y. float4-vectorized.
// ---------------------------------------------------------------------------
__global__ __launch_bounds__(256) void gate_norm_kernel(
    const float* __restrict__ proj, const float* __restrict__ norm_w,
    float* __restrict__ y) {
  const int s = blockIdx.x;
  const int tid = threadIdx.x;
  __shared__ float red[256];
  float4 fv[4];
  float ss = 0.f;
#pragma unroll
  for (int i = 0; i < 4; ++i) {
    int ch = tid * 16 + i * 4;
    float4 g4 = *(const float4*)&proj[(size_t)s * PROJ + ch];
    float4 y4 = *(const float4*)&y[(size_t)s * INTER + ch];
    float4 f;
    f.x = y4.x * (g4.x / (1.f + __expf(-g4.x)));
    f.y = y4.y * (g4.y / (1.f + __expf(-g4.y)));
    f.z = y4.z * (g4.z / (1.f + __expf(-g4.z)));
    f.w = y4.w * (g4.w / (1.f + __expf(-g4.w)));
    fv[i] = f;
    ss += f.x * f.x + f.y * f.y + f.z * f.z + f.w * f.w;
  }
  red[tid] = ss;
  __syncthreads();
  for (int off = 128; off > 0; off >>= 1) {
    if (tid < off) red[tid] += red[tid + off];
    __syncthreads();
  }
  const float scale = rsqrtf(red[0] / (float)INTER + EPS);
#pragma unroll
  for (int i = 0; i < 4; ++i) {
    int ch = tid * 16 + i * 4;
    float4 w4 = *(const float4*)&norm_w[ch];
    float4 o;
    o.x = w4.x * fv[i].x * scale;
    o.y = w4.y * fv[i].y * scale;
    o.z = w4.z * fv[i].z * scale;
    o.w = w4.w * fv[i].w * scale;
    *(float4*)&y[(size_t)s * INTER + ch] = o;
  }
}

// ---------------------------------------------------------------------------
// Launch
// ---------------------------------------------------------------------------
extern "C" void kernel_launch(void* const* d_in, const int* in_sizes, int n_in,
                              void* d_out, int out_size, void* d_ws, size_t ws_size,
                              hipStream_t stream) {
  const float* X       = (const float*)d_in[0];  // [S,E]
  // d_in[1] = mask (unused by reference)
  const float* W_in    = (const float*)d_in[2];  // [PROJ,E]
  const float* conv_w  = (const float*)d_in[3];  // [CONV_DIM,4]
  const float* conv_b  = (const float*)d_in[4];  // [CONV_DIM]
  const float* dt_bias = (const float*)d_in[5];  // [NH]
  const float* A_log   = (const float*)d_in[6];  // [NH]
  const float* Dvec    = (const float*)d_in[7];  // [NH]
  const float* norm_w  = (const float*)d_in[8];  // [INTER]
  const float* W_out   = (const float*)d_in[9];  // [E,INTER]
  float* out = (float*)d_out;

  // Workspace layout: proj fp32 (69.7MB) | convb bf16 (17.8MB) | dt/dA | y fp32
  float*  proj  = (float*)d_ws;                          // S*PROJ fp32
  bf16_t* convb = (bf16_t*)(proj + (size_t)S_ * PROJ);   // S*CONV_DIM bf16
  float*  dtsp  = (float*)(convb + (size_t)S_ * CONV_DIM);
  float*  dAv   = dtsp + (size_t)S_ * NH;
  float*  ybuf  = dAv + (size_t)S_ * NH;                 // S*INTER fp32

  // 1) proj = X * W_in^T        [2048 x 8512, K=2048]
  gemm_nt_wmma<<<dim3(PROJ / 64, S_ / 256), 256, 0, stream>>>(
      X, W_in, proj, S_, PROJ, E_);

  // 2) causal depthwise conv + SiLU over h_BC channels (bf16 out)
  conv_silu_kernel<<<(S_ * (CONV_DIM / 4)) / 256, 256, 0, stream>>>(
      proj, conv_w, conv_b, convb);

  // 3) dt softplus + dA
  dt_kernel<<<(S_ * NH) / 256, 256, 0, stream>>>(proj, dt_bias, A_log, dtsp, dAv);

  // 4) chunked SSD core (WMMA x2 per tile, async-to-LDS staging)
  ssd_kernel<<<dim3(CS / 64, NH, NCHUNK), 256, 0, stream>>>(
      convb, dtsp, dAv, Dvec, ybuf);

  // 5) gate + RMSNorm (in place on ybuf)
  gate_norm_kernel<<<S_, 256, 0, stream>>>(proj, norm_w, ybuf);

  // 6) out = yn * W_out^T       [2048 x 2048, K=4096]
  gemm_nt_wmma<<<dim3(E_ / 64, S_ / 256), 256, 0, stream>>>(
      ybuf, W_out, out, S_, E_, INTER);
}
